// CGLSTM_37830071943202
// MI455X (gfx1250) — compile-verified, matrix operand order checked by reading, not run on hardware
//
#include <hip/hip_runtime.h>

#define B_ 256
#define T_ 512
#define D_ 200
#define H_ 128
#define F_ 128

typedef __attribute__((ext_vector_type(16))) __bf16 v16bf;
typedef __attribute__((ext_vector_type(8)))  float  v8f;
typedef __attribute__((ext_vector_type(4)))  unsigned int v4u;
typedef __attribute__((ext_vector_type(8)))  int v8i;
typedef __attribute__((ext_vector_type(4)))  int v4i;

__device__ __forceinline__ float fast_sigmoid(float x) {
    return 1.0f / (1.0f + __expf(-x));
}
__device__ __forceinline__ float fast_tanh(float x) {
    float ax = fabsf(x);
    float e = __expf(-2.0f * ax);
    float t = (1.0f - e) / (1.0f + e);
    return copysignf(t, x);
}

// ---- WMMA fragment helpers (16x16x32 bf16, wave32) -------------------------
// A fragment: lane L<16 -> row M=L, K in {0..7, 16..23}; L>=16 -> row M=L-16,
// K in {8..15, 24..31}. Two contiguous 16B chunks per lane -> ds_load_b128 x2.
__device__ __forceinline__ v16bf load_a_frag(const __bf16* p, int ld) {
    int lane = threadIdx.x & 31;
    const __bf16* r = p + (lane & 15) * ld + ((lane >> 4) << 3);
    v16bf a;
#pragma unroll
    for (int i = 0; i < 8; ++i) a[i] = r[i];
#pragma unroll
    for (int i = 0; i < 8; ++i) a[8 + i] = r[16 + i];
    return a;
}

// B fragments are pre-packed: tile (kg,nt) stored as 32 lanes x 16 contiguous
// bf16 -> each lane does one 32B contiguous global load.
__device__ __forceinline__ v16bf load_b_packed(const __bf16* wp, int kg, int nt, int nTiles) {
    int lane = threadIdx.x & 31;
    return *(const v16bf*)(wp + (((size_t)(kg * nTiles + nt) * 32 + lane) << 4));
}

// One 16x16 output tile accumulated over kGroups k-slices of 32.
__device__ __forceinline__ v8f gemm_tile(const __bf16* A, int ldA, const __bf16* Wp,
                                         int nt, int nTiles, int kGroups, v8f acc) {
    for (int kg = 0; kg < kGroups; ++kg) {
        v16bf a = load_a_frag(A + kg * 32, ldA);
        v16bf b = load_b_packed(Wp, kg, nt, nTiles);
        acc = __builtin_amdgcn_wmma_f32_16x16x32_bf16(false, a, false, b,
                                                      (short)0, acc, false, false);
    }
    return acc;
}

// ---- TDM: async 2D tile load (16 rows x 512 f32, row stride T*512) ---------
// Descriptor packing per CDNA5 ISA.
// This toolchain exposes the 6-arg builtin (g0, g1, g2, g3, g1-ext, cpol).
__device__ __forceinline__ void tdm_load_2d(unsigned ldsOff, const float* gaddr) {
    unsigned long long ga = (unsigned long long)(__UINTPTR_TYPE__)gaddr;
    v4u g0;
    g0[0] = 1u;                                            // count=1, user mode
    g0[1] = ldsOff;                                        // lds_addr (bytes)
    g0[2] = (unsigned)(ga & 0xffffffffu);                  // global_addr[31:0]
    g0[3] = (unsigned)((ga >> 32) & 0x1ffffffu) | (2u << 30); // addr[56:32] | type=2
    v8i g1;
    g1[0] = (2 << 16);            // workgroup_mask=0, data_size=2 (4 bytes)
    g1[1] = (512 << 16);          // tensor_dim0 = 512 (bits 79:48, low part)
    g1[2] = 0;                    // tensor_dim0 hi=0 | tensor_dim1 lo=0
    g1[3] = 2 | (512 << 16);      // tensor_dim1 hi (131072>>16) | tile_dim0=512
    g1[4] = 16;                   // tile_dim1 = 16 rows, tile_dim2 = 0
    g1[5] = (int)(T_ * 512);      // tensor_dim0_stride (elements), low 32
    g1[6] = 0;                    // stride hi | tensor_dim1_stride lo
    g1[7] = 0;
    v4i z4 = {0, 0, 0, 0};
    v8i z8 = {0, 0, 0, 0, 0, 0, 0, 0};
    __builtin_amdgcn_tensor_load_to_lds(g0, g1, z4, z4, z8, 0);
}

// ---- weight packing --------------------------------------------------------
// src is [K,N] row-major (row stride = rowStride); rows >= kMax are zero pad.
__global__ void pack_b_kernel(const float* __restrict__ src, __bf16* __restrict__ dst,
                              int N, int rowStride, int kMax, float scale) {
    int nTiles = N >> 4;
    int tile = blockIdx.x;
    int kg = tile / nTiles, nt = tile - kg * nTiles;
    int lane = threadIdx.x;
    int n = nt * 16 + (lane & 15);
    int koff = kg * 32 + ((lane >> 4) << 3);
    __bf16* out = dst + (((size_t)tile * 32 + lane) << 4);
#pragma unroll
    for (int i = 0; i < 8; ++i) {
        int k = koff + i;
        out[i] = (__bf16)((k < kMax) ? src[(size_t)k * rowStride + n] * scale : 0.0f);
    }
#pragma unroll
    for (int i = 0; i < 8; ++i) {
        int k = koff + 16 + i;
        out[8 + i] = (__bf16)((k < kMax) ? src[(size_t)k * rowStride + n] * scale : 0.0f);
    }
}

// gam_w layout [4][128][128] gate-major -> Geff [K=128, N=512] with gate signs
// (+1, -LAM, +1, +1) folded in.
__global__ void pack_gamma_kernel(const float* __restrict__ gw, __bf16* __restrict__ dst) {
    const int nTiles = 32;
    int tile = blockIdx.x;                 // 4 kGroups * 32 nTiles = 128
    int kg = tile / nTiles, nt = tile - kg * nTiles;
    int lane = threadIdx.x;
    int n = nt * 16 + (lane & 15);
    int g = n >> 7, col = n & 127;
    float sgn = (g == 1) ? -0.5f : 1.0f;   // LAM = 0.5 folded into f-gate weights
    int koff = kg * 32 + ((lane >> 4) << 3);
    __bf16* out = dst + (((size_t)tile * 32 + lane) << 4);
#pragma unroll
    for (int i = 0; i < 8; ++i) {
        int k = koff + i;
        out[i] = (__bf16)(sgn * gw[((size_t)g * 128 + k) * 128 + col]);
    }
#pragma unroll
    for (int i = 0; i < 8; ++i) {
        int k = koff + 16 + i;
        out[8 + i] = (__bf16)(sgn * gw[((size_t)g * 128 + k) * 128 + col]);
    }
}

// ---- TwoStageFusion: gamma_sh[b,t] = shifted gamma, bf16 -------------------
__global__ __launch_bounds__(128) void fusion_kernel(
    const float* __restrict__ ampc, const float* __restrict__ phc,
    const float* __restrict__ rlos,
    const float* __restrict__ f_amp_w, const float* __restrict__ f_amp_b,
    const float* __restrict__ f_ph_w,  const float* __restrict__ f_ph_b,
    const float* __restrict__ f_gate_b,
    const float* __restrict__ f_rlos_w1, const float* __restrict__ f_rlos_b1,
    const float* __restrict__ f_rlos_b2,
    const float* __restrict__ f_q_b, const float* __restrict__ f_k_b,
    const float* __restrict__ f_v_b, const float* __restrict__ f_out_b,
    const __bf16* __restrict__ Wg_p, const __bf16* __restrict__ Wrl2_p,
    const __bf16* __restrict__ Wq_p, const __bf16* __restrict__ Wk_p,
    const __bf16* __restrict__ Wv_p, const __bf16* __restrict__ Wo_p,
    __bf16* __restrict__ gamma_sh)
{
    __shared__ __bf16 sCat[16][256];   // [ph | amp]
    __shared__ __bf16 sRl1[16][128];
    __shared__ __bf16 sCorr[16][128];
    __shared__ __bf16 sRl[16][128];
    __shared__ float  sQ[16][128];
    __shared__ float  sK[16][128];
    __shared__ __bf16 sSV[16][128];
    __shared__ float  sS[16];

    const int r0 = blockIdx.x * 16;    // flat (b*T + t0), t0 aligned to 16
    const int tid = threadIdx.x;
    const int wave = tid >> 5;
    const int lane = tid & 31;

    if ((r0 & (T_ - 1)) == 0) {        // gamma at t=0 is zeros
        gamma_sh[(size_t)r0 * F_ + tid] = (__bf16)0.0f;
    }

    // elementwise: amp, ph, rl1
    for (int e = tid; e < 16 * 128; e += 128) {
        int m = e >> 7, n = e & 127;
        int r = r0 + m;
        float av = fast_tanh(ampc[r] * f_amp_w[n] + f_amp_b[n]);
        float pv = fast_tanh(phc[r]  * f_ph_w[n]  + f_ph_b[n]);
        sCat[m][n] = (__bf16)pv;
        sCat[m][128 + n] = (__bf16)av;
        float d0 = rlos[(size_t)r * 2 + 0], d1 = rlos[(size_t)r * 2 + 1];
        sRl1[m][n] = (__bf16)fast_tanh(d0 * f_rlos_w1[n] + d1 * f_rlos_w1[128 + n] + f_rlos_b1[n]);
    }
    __syncthreads();

    // beta = sigmoid([ph|amp] @ Wg + b) ; corr = beta*ph + (1-beta)*amp
    for (int q = 0; q < 2; ++q) {
        int nt = wave * 2 + q;
        v8f acc = {};
        acc = gemm_tile(&sCat[0][0], 256, Wg_p, nt, 8, 8, acc);
        int n = nt * 16 + (lane & 15);
        int mb = (lane >> 4) << 3;
#pragma unroll
        for (int j = 0; j < 8; ++j) {
            int m = mb + j;
            float beta = fast_sigmoid(acc[j] + f_gate_b[n]);
            float pv = (float)sCat[m][n];
            float av = (float)sCat[m][128 + n];
            sCorr[m][n] = (__bf16)(beta * pv + (1.0f - beta) * av);
        }
    }
    __syncthreads();

    // rl = tanh(rl1 @ W2 + b2)
    for (int q = 0; q < 2; ++q) {
        int nt = wave * 2 + q;
        v8f acc = {};
        acc = gemm_tile(&sRl1[0][0], 128, Wrl2_p, nt, 8, 4, acc);
        int n = nt * 16 + (lane & 15);
        int mb = (lane >> 4) << 3;
#pragma unroll
        for (int j = 0; j < 8; ++j)
            sRl[mb + j][n] = (__bf16)fast_tanh(acc[j] + f_rlos_b2[n]);
    }
    __syncthreads();

    // q = corr@Wq + bq ; k = rl@Wk + bk
    for (int q = 0; q < 2; ++q) {
        int nt = wave * 2 + q;
        v8f aq = {}; aq = gemm_tile(&sCorr[0][0], 128, Wq_p, nt, 8, 4, aq);
        v8f ak = {}; ak = gemm_tile(&sRl[0][0],   128, Wk_p, nt, 8, 4, ak);
        int n = nt * 16 + (lane & 15);
        int mb = (lane >> 4) << 3;
#pragma unroll
        for (int j = 0; j < 8; ++j) {
            sQ[mb + j][n] = aq[j] + f_q_b[n];
            sK[mb + j][n] = ak[j] + f_k_b[n];
        }
    }
    __syncthreads();

    if (tid < 16) {
        float s = 0.0f;
        for (int n = 0; n < 128; ++n) s += sQ[tid][n] * sK[tid][n];
        sS[tid] = fast_sigmoid(s * 0.08838834764831845f);   // F^-0.5
    }
    __syncthreads();

    // sv = attn * (rl@Wv + bv)
    for (int q = 0; q < 2; ++q) {
        int nt = wave * 2 + q;
        v8f acc = {};
        acc = gemm_tile(&sRl[0][0], 128, Wv_p, nt, 8, 4, acc);
        int n = nt * 16 + (lane & 15);
        int mb = (lane >> 4) << 3;
#pragma unroll
        for (int j = 0; j < 8; ++j) {
            int m = mb + j;
            sSV[m][n] = (__bf16)((acc[j] + f_v_b[n]) * sS[m]);
        }
    }
    __syncthreads();

    // gamma = sv @ Wo + bo, stored shifted by one step (row r -> r+1)
    for (int q = 0; q < 2; ++q) {
        int nt = wave * 2 + q;
        v8f acc = {};
        acc = gemm_tile(&sSV[0][0], 128, Wo_p, nt, 8, 4, acc);
        int n = nt * 16 + (lane & 15);
        int mb = (lane >> 4) << 3;
#pragma unroll
        for (int j = 0; j < 8; ++j) {
            int r = r0 + mb + j;
            if ((r & (T_ - 1)) != T_ - 1)
                gamma_sh[(size_t)(r + 1) * F_ + n] = (__bf16)(acc[j] + f_out_b[n]);
        }
    }
}

// ---- Precompute gate terms Pl[b,t,4H] = (x-part) + bias + gamma@Geff -------
__global__ __launch_bounds__(256) void precompute_kernel(
    const float* __restrict__ hrrp,
    const __bf16* __restrict__ gamma_sh,
    const float* __restrict__ base_b0, const float* __restrict__ base_b12,
    const __bf16* __restrict__ W0x_p,
    const __bf16* __restrict__ Geff0_p, const __bf16* __restrict__ Geff1_p,
    const __bf16* __restrict__ Geff2_p,
    float* __restrict__ P0, float* __restrict__ P1, float* __restrict__ P2)
{
    __shared__ __bf16 sX[16][224];     // |hrrp| padded K 200 -> 224
    __shared__ __bf16 sG[16][128];
    const int r0 = blockIdx.x * 16;
    const int layer = blockIdx.y;
    const int tid = threadIdx.x;
    const int wave = tid >> 5;
    const int lane = tid & 31;

    for (int e = tid; e < 16 * 128; e += 256) {
        int m = e >> 7, n = e & 127;
        sG[m][n] = gamma_sh[(size_t)(r0 + m) * F_ + n];
    }
    if (layer == 0) {
        for (int e = tid; e < 16 * 224; e += 256) {
            int m = e / 224, k = e - m * 224;
            sX[m][k] = (__bf16)((k < D_) ? fabsf(hrrp[(size_t)(r0 + m) * D_ + k]) : 0.0f);
        }
    }
    __syncthreads();

    const __bf16* Geff = (layer == 0) ? Geff0_p : (layer == 1 ? Geff1_p : Geff2_p);
    const float* bias  = (layer == 0) ? base_b0 : (base_b12 + (size_t)(layer - 1) * 512);
    float* P = (layer == 0) ? P0 : (layer == 1 ? P1 : P2);

    for (int q = 0; q < 4; ++q) {
        int nt = wave * 4 + q;   // 0..31
        v8f acc = {};
        if (layer == 0)
            acc = gemm_tile(&sX[0][0], 224, W0x_p, nt, 32, 7, acc);
        acc = gemm_tile(&sG[0][0], 128, Geff, nt, 32, 4, acc);
        int n = nt * 16 + (lane & 15);
        int mb = (lane >> 4) << 3;
#pragma unroll
        for (int j = 0; j < 8; ++j) {
            int r = r0 + mb + j;
            P[(size_t)r * 512 + n] = acc[j] + bias[n];
        }
    }
}

// ---- Persistent recurrent scan: 16 WGs, each owns a 16-batch slice ---------
// Dynamic LDS layout (233472 bytes total):
//   [0      .. 12288)  sH[3][16][128] bf16
//   [12288  .. 36864)  sC[3][16][128] f32
//   [36864  .. 233472) sP[2][3][16][512] f32  (TDM double buffer)
#define SP_OFF 36864u
__global__ __launch_bounds__(256) void scan_kernel(
    const float* __restrict__ P0, const float* __restrict__ P1, const float* __restrict__ P2,
    const __bf16* __restrict__ W0h_p,
    const __bf16* __restrict__ W1h_p, const __bf16* __restrict__ W1x_p,
    const __bf16* __restrict__ W2h_p, const __bf16* __restrict__ W2x_p,
    const float* __restrict__ reg_w, const float* __restrict__ reg_b,
    float* __restrict__ out)
{
    extern __shared__ char smem[];
    __bf16* sH = (__bf16*)smem;                 // [3][16][128]
    float*  sC = (float*)(smem + 12288);        // [3][16][128]
    float*  sP = (float*)(smem + SP_OFF);       // [2][3][16][512]

    const int bBase = blockIdx.x * 16;
    const int tid = threadIdx.x;
    const int wave = tid >> 5;           // 0..7 -> owns columns [16w,16w+16)
    const int lane = tid & 31;

    for (int e = tid; e < 3 * 16 * 128; e += 256) {
        sH[e] = (__bf16)0.0f;
        sC[e] = 0.0f;
    }

    // TDM: stage the three 16x512 f32 gate-term tiles for step t into buf.
    auto issue_P = [&](int t, int buf) {
        if (wave == 0) {
            size_t rowbase = ((size_t)bBase * T_ + t) * 512;
            unsigned lbase = SP_OFF + (unsigned)buf * (3u * 16u * 512u * 4u);
            tdm_load_2d(lbase + 0u * 32768u, P0 + rowbase);
            tdm_load_2d(lbase + 1u * 32768u, P1 + rowbase);
            tdm_load_2d(lbase + 2u * 32768u, P2 + rowbase);
        }
    };

    issue_P(0, 0);
    if (wave == 0) __builtin_amdgcn_s_wait_tensorcnt(0);
    __syncthreads();

    // per layer: wave w computes gate tiles nt = {w, w+8, w+16, w+24}, i.e. the
    // SAME 16 columns of i/f/c/o -> full LSTM pointwise math stays in registers
    auto do_layer = [&](int l, const __bf16* Wh_p, const __bf16* Wx_p,
                        const float* __restrict__ Pl) {
        v8f acc[4] = {};
#pragma unroll
        for (int g = 0; g < 4; ++g) {
            int nt = g * 8 + wave;
            acc[g] = gemm_tile(sH + l * (16 * 128), 128, Wh_p, nt, 32, 4, acc[g]);
            if (Wx_p)
                acc[g] = gemm_tile(sH + (l - 1) * (16 * 128), 128, Wx_p, nt, 32, 4, acc[g]);
        }
        __syncthreads();                 // all reads of sH done before writes
        int n = wave * 16 + (lane & 15);
        int mb = (lane >> 4) << 3;
#pragma unroll
        for (int j = 0; j < 8; ++j) {
            int m = mb + j;
            int base = m * 512 + n;
            float i  = fast_sigmoid(acc[0][j] + Pl[base]);
            float f  = fast_sigmoid(acc[1][j] + Pl[base + 128]);
            float ch = fast_tanh  (acc[2][j] + Pl[base + 256]);
            float o  = fast_sigmoid(acc[3][j] + Pl[base + 384]);
            float c = f * sC[l * (16 * 128) + m * 128 + n] + i * ch;
            sC[l * (16 * 128) + m * 128 + n] = c;
            sH[l * (16 * 128) + m * 128 + n] = (__bf16)(o * fast_tanh(c));
        }
        __syncthreads();
    };

    for (int t = 0; t < T_; ++t) {
        int buf = t & 1;
        if (t + 1 < T_) issue_P(t + 1, buf ^ 1);   // overlap DMA with compute
        const float* Pb = sP + buf * (3 * 16 * 512);
        do_layer(0, W0h_p, nullptr, Pb + 0 * (16 * 512));
        do_layer(1, W1h_p, W1x_p,  Pb + 1 * (16 * 512));
        do_layer(2, W2h_p, W2x_p,  Pb + 2 * (16 * 512));
        if (t + 1 < T_) {
            if (wave == 0) __builtin_amdgcn_s_wait_tensorcnt(0);
            __syncthreads();             // next-step tiles resident for all waves
        }
    }

    if (tid < 32) {                      // out = h2 @ reg_w + reg_b
        int m = tid >> 1, o = tid & 1;
        float s = reg_b[o];
        for (int n = 0; n < 128; ++n)
            s += (float)sH[2 * (16 * 128) + m * 128 + n] * reg_w[n * 2 + o];
        out[(size_t)(bBase + m) * 2 + o] = s;
    }
}

// ---------------------------------------------------------------------------
extern "C" void kernel_launch(void* const* d_in, const int* in_sizes, int n_in,
                              void* d_out, int out_size, void* d_ws, size_t ws_size,
                              hipStream_t stream) {
    (void)in_sizes; (void)n_in; (void)out_size; (void)ws_size;
    const float* hrrp     = (const float*)d_in[0];
    const float* ampc     = (const float*)d_in[1];
    const float* phc      = (const float*)d_in[2];
    const float* rlos     = (const float*)d_in[3];
    const float* f_amp_w  = (const float*)d_in[4];
    const float* f_amp_b  = (const float*)d_in[5];
    const float* f_ph_w   = (const float*)d_in[6];
    const float* f_ph_b   = (const float*)d_in[7];
    const float* f_gate_w = (const float*)d_in[8];
    const float* f_gate_b = (const float*)d_in[9];
    const float* f_rl_w1  = (const float*)d_in[10];
    const float* f_rl_b1  = (const float*)d_in[11];
    const float* f_rl_w2  = (const float*)d_in[12];
    const float* f_rl_b2  = (const float*)d_in[13];
    const float* f_q_w    = (const float*)d_in[14];
    const float* f_q_b    = (const float*)d_in[15];
    const float* f_k_w    = (const float*)d_in[16];
    const float* f_k_b    = (const float*)d_in[17];
    const float* f_v_w    = (const float*)d_in[18];
    const float* f_v_b    = (const float*)d_in[19];
    const float* f_out_w  = (const float*)d_in[20];
    const float* f_out_b  = (const float*)d_in[21];
    const float* base_w0  = (const float*)d_in[22];
    const float* base_b0  = (const float*)d_in[23];
    const float* gam_w0   = (const float*)d_in[24];
    const float* base_w12 = (const float*)d_in[25];
    const float* base_b12 = (const float*)d_in[26];
    const float* gam_w12  = (const float*)d_in[27];
    const float* reg_w    = (const float*)d_in[28];
    const float* reg_b    = (const float*)d_in[29];

    char* ws = (char*)d_ws;
    size_t off = 0;
    auto carve = [&](size_t bytes) -> char* {
        char* p = ws + off;
        off += (bytes + 255) & ~(size_t)255;
        return p;
    };
    __bf16* Wg_p   = (__bf16*)carve((size_t)256 * 128 * 2);
    __bf16* Wrl2_p = (__bf16*)carve((size_t)128 * 128 * 2);
    __bf16* Wq_p   = (__bf16*)carve((size_t)128 * 128 * 2);
    __bf16* Wk_p   = (__bf16*)carve((size_t)128 * 128 * 2);
    __bf16* Wv_p   = (__bf16*)carve((size_t)128 * 128 * 2);
    __bf16* Wo_p   = (__bf16*)carve((size_t)128 * 128 * 2);
    __bf16* W0x_p  = (__bf16*)carve((size_t)224 * 512 * 2);
    __bf16* W0h_p  = (__bf16*)carve((size_t)128 * 512 * 2);
    __bf16* W1h_p  = (__bf16*)carve((size_t)128 * 512 * 2);
    __bf16* W1x_p  = (__bf16*)carve((size_t)128 * 512 * 2);
    __bf16* W2h_p  = (__bf16*)carve((size_t)128 * 512 * 2);
    __bf16* W2x_p  = (__bf16*)carve((size_t)128 * 512 * 2);
    __bf16* G0_p   = (__bf16*)carve((size_t)128 * 512 * 2);
    __bf16* G1_p   = (__bf16*)carve((size_t)128 * 512 * 2);
    __bf16* G2_p   = (__bf16*)carve((size_t)128 * 512 * 2);
    __bf16* gamma  = (__bf16*)carve((size_t)B_ * T_ * F_ * 2);
    float*  P0     = (float*) carve((size_t)B_ * T_ * 512 * 4);
    float*  P1     = (float*) carve((size_t)B_ * T_ * 512 * 4);
    float*  P2     = (float*) carve((size_t)B_ * T_ * 512 * 4);

    dim3 b32(32);
    // pack fusion weights ([K,N] row-major -> fragment layout, bf16)
    pack_b_kernel<<<(256 / 32) * (128 / 16), b32, 0, stream>>>(f_gate_w, Wg_p, 128, 128, 256, 1.0f);
    pack_b_kernel<<<(128 / 32) * (128 / 16), b32, 0, stream>>>(f_rl_w2, Wrl2_p, 128, 128, 128, 1.0f);
    pack_b_kernel<<<(128 / 32) * (128 / 16), b32, 0, stream>>>(f_q_w,   Wq_p,  128, 128, 128, 1.0f);
    pack_b_kernel<<<(128 / 32) * (128 / 16), b32, 0, stream>>>(f_k_w,   Wk_p,  128, 128, 128, 1.0f);
    pack_b_kernel<<<(128 / 32) * (128 / 16), b32, 0, stream>>>(f_v_w,   Wv_p,  128, 128, 128, 1.0f);
    pack_b_kernel<<<(128 / 32) * (128 / 16), b32, 0, stream>>>(f_out_w, Wo_p,  128, 128, 128, 1.0f);
    // base_w0 [328,512]: rows 0..127 = h-part, rows 128..327 = x-part (pad K to 224)
    pack_b_kernel<<<(224 / 32) * (512 / 16), b32, 0, stream>>>(base_w0 + (size_t)128 * 512, W0x_p, 512, 512, 200, 1.0f);
    pack_b_kernel<<<(128 / 32) * (512 / 16), b32, 0, stream>>>(base_w0, W0h_p, 512, 512, 128, 1.0f);
    // base_w12 [2][256,512]: per layer rows 0..127 = h, 128..255 = x(=lower h)
    pack_b_kernel<<<(128 / 32) * (512 / 16), b32, 0, stream>>>(base_w12,                      W1h_p, 512, 512, 128, 1.0f);
    pack_b_kernel<<<(128 / 32) * (512 / 16), b32, 0, stream>>>(base_w12 + (size_t)128 * 512,  W1x_p, 512, 512, 128, 1.0f);
    pack_b_kernel<<<(128 / 32) * (512 / 16), b32, 0, stream>>>(base_w12 + (size_t)256 * 512,  W2h_p, 512, 512, 128, 1.0f);
    pack_b_kernel<<<(128 / 32) * (512 / 16), b32, 0, stream>>>(base_w12 + (size_t)384 * 512,  W2x_p, 512, 512, 128, 1.0f);
    // gamma-modulation weights with gate signs folded (i:+1, f:-LAM, c:+1, o:+1)
    pack_gamma_kernel<<<128, b32, 0, stream>>>(gam_w0, G0_p);
    pack_gamma_kernel<<<128, b32, 0, stream>>>(gam_w12, G1_p);
    pack_gamma_kernel<<<128, b32, 0, stream>>>(gam_w12 + (size_t)4 * 128 * 128, G2_p);

    // fused TwoStageFusion -> shifted gamma (bf16)
    fusion_kernel<<<(B_ * T_) / 16, 128, 0, stream>>>(
        ampc, phc, rlos, f_amp_w, f_amp_b, f_ph_w, f_ph_b, f_gate_b,
        f_rl_w1, f_rl_b1, f_rl_b2, f_q_b, f_k_b, f_v_b, f_out_b,
        Wg_p, Wrl2_p, Wq_p, Wk_p, Wv_p, Wo_p, gamma);

    // precompute per-layer non-recurrent gate terms
    precompute_kernel<<<dim3((B_ * T_) / 16, 3), 256, 0, stream>>>(
        hrrp, gamma, base_b0, base_b12, W0x_p, G0_p, G1_p, G2_p, P0, P1, P2);

    // persistent recurrent scan + regression head (TDM double-buffered P)
    const size_t scan_lds = 36864 + (size_t)2 * 3 * 16 * 512 * 4;   // 233472 B
    scan_kernel<<<B_ / 16, 256, scan_lds, stream>>>(
        P0, P1, P2, W0h_p, W1h_p, W1x_p, W2h_p, W2x_p, reg_w, reg_b, (float*)d_out);
}
